// HGC_CPT_54932631715894
// MI455X (gfx1250) — matrix operand order repaired and checked by reading, not compiled, not run on hardware
//
#include <hip/hip_runtime.h>
#include <hip/hip_bf16.h>

typedef __attribute__((ext_vector_type(16))) __bf16 v16bf;
typedef __attribute__((ext_vector_type(8)))  __bf16 v8bf;
typedef __attribute__((ext_vector_type(8)))  float  v8f;

static constexpr int NN = 20000;   // nodes
static constexpr int MM = 1024;    // init columns
static constexpr int EE = 640000;  // edges per branch
static constexpr int DD = 256;     // feature dim
static constexpr int HH = 16;      // MLP hidden

// ---------------------------------------------------------------------------
// Column-embedding MLP: colT[d][m] = (ReLU(m*W1+b1) @ W2 + b2)[d]  (transposed,
// bf16) so it can serve directly as the n-major B operand of the WMMA GEMM.
// ---------------------------------------------------------------------------
__global__ void k_col_proj(const float* __restrict__ pW1, const float* __restrict__ pb1,
                           const float* __restrict__ pW2, const float* __restrict__ pb2,
                           __bf16* __restrict__ colT) {
    int m = blockIdx.x * blockDim.x + threadIdx.x;
    if (m >= MM) return;
    float h[HH];
    float x = (float)m;
#pragma unroll
    for (int i = 0; i < HH; ++i) {
        float v = x * pW1[i] + pb1[i];
        h[i] = v > 0.f ? v : 0.f;
    }
    for (int d = 0; d < DD; ++d) {
        float acc = pb2[d];
#pragma unroll
        for (int i = 0; i < HH; ++i) acc += h[i] * pW2[i * DD + d];
        colT[(size_t)d * MM + m] = (__bf16)acc;
    }
}

// maskbf[i] = (init[i] != 0) in bf16 ({0,1} is bf16-exact)
__global__ void k_mask(const float* __restrict__ init, __bf16* __restrict__ maskbf, int total) {
    int i = blockIdx.x * blockDim.x + threadIdx.x;
    if (i < total) maskbf[i] = (init[i] != 0.f) ? (__bf16)1.0f : (__bf16)0.0f;
}

// one wave per row: rinv[n] = cnt>0 ? 1/cnt : 0
__global__ void k_rowcnt(const __bf16* __restrict__ maskbf, float* __restrict__ rinv) {
    int w    = (blockIdx.x * blockDim.x + threadIdx.x) >> 5;
    int lane = threadIdx.x & 31;
    if (w >= NN) return;
    const __bf16* row = maskbf + (size_t)w * MM;
    float s = 0.f;
    for (int j = lane; j < MM; j += 32) s += (float)row[j];
#pragma unroll
    for (int off = 16; off; off >>= 1) s += __shfl_xor(s, off, 32);
    if (lane == 0) rinv[w] = s > 0.f ? 1.f / s : 0.f;
}

// ---------------------------------------------------------------------------
// Generic bf16 WMMA GEMM:  C[Nr, NC] = A[Nr, K] * Bt^T   (Bt is n-major [NC,K])
// One wave per 16x16 output tile, K stepped by 32 (v_wmma_f32_16x16x32_bf16).
// ---------------------------------------------------------------------------
__global__ void __launch_bounds__(256)
k_wmma_gemm(const __bf16* __restrict__ A, const __bf16* __restrict__ Bt,
            float* __restrict__ C, int K, int NC, int total_tiles) {
    int w = (blockIdx.x * blockDim.x + threadIdx.x) >> 5;
    if (w >= total_tiles) return;                 // wave-uniform: EXEC stays all-1s
    int lane = threadIdx.x & 31;
    int ntc  = NC >> 4;
    int tm   = w / ntc;
    int tn   = w - tm * ntc;
    int g    = lane >> 4;                         // lane group (K-half select)
    int r    = lane & 15;                         // row (A) / col (B) within tile

    const __bf16* Arow = A  + (size_t)(tm * 16 + r) * K;
    const __bf16* Brow = Bt + (size_t)(tn * 16 + r) * K;

    v8f c = {};
    for (int k0 = 0; k0 < K; k0 += 32) {
        // A 16x32 bf16 fragment: lane holds K in [8g,8g+8) and [16+8g,16+8g+8)
        v8bf a0 = *(const v8bf*)(Arow + k0 + 8 * g);
        v8bf a1 = *(const v8bf*)(Arow + k0 + 16 + 8 * g);
        // B 32x16 bf16 fragment: lane holds K in [8g,8g+8) and [16+8g,16+8g+8), col = lane&15
        v8bf b0 = *(const v8bf*)(Brow + k0 + 8 * g);
        v8bf b1 = *(const v8bf*)(Brow + k0 + 16 + 8 * g);
        v16bf a, b;
#pragma unroll
        for (int j = 0; j < 8; ++j) {
            a[j] = a0[j]; a[j + 8] = a1[j];
            b[j] = b0[j]; b[j + 8] = b1[j];
        }
        c = __builtin_amdgcn_wmma_f32_16x16x32_bf16(false, a, false, b,
                                                    (short)0, c, false, false);
    }
    // D layout: element v -> row 8g+v, col r
    float* Cout = C + (size_t)(tm * 16 + 8 * g) * NC + tn * 16 + r;
#pragma unroll
    for (int v = 0; v < 8; ++v) Cout[(size_t)v * NC] = c[v];
}

// x0bf = bf16( gemmC * rinv[row] )   (masked mean; zero rows stay zero)
__global__ void k_scale_bf(const float* __restrict__ C, const float* __restrict__ rinv,
                           __bf16* __restrict__ xbf, int total) {
    int i = blockIdx.x * blockDim.x + threadIdx.x;
    if (i >= total) return;
    int n = i / DD;
    xbf[i] = (__bf16)(C[i] * rinv[n]);
}

// Wt[n*DD + k] = bf16(W[k*DD + n])   (transpose + downconvert one layer weight)
__global__ void k_wt(const float* __restrict__ W, __bf16* __restrict__ Wt) {
    int i = blockIdx.x * blockDim.x + threadIdx.x;
    if (i >= DD * DD) return;
    int n = i / DD, k = i - n * DD;
    Wt[i] = (__bf16)W[k * DD + n];
}

__global__ void k_zero(float* __restrict__ p, int n) {
    int i = blockIdx.x * blockDim.x + threadIdx.x;
    if (i < n) p[i] = 0.f;
}

// deg accumulation over edge destinations
__global__ void k_deg(const int* __restrict__ col, const float* __restrict__ w,
                      float* __restrict__ deg) {
    int e = blockIdx.x * blockDim.x + threadIdx.x;
    if (e < EE) atomicAdd(&deg[col[e]], w[e]);
}

// dinv[i] = rsqrt(deg + 1) (self loop weight 1; deg+1 > 0 always)
__global__ void k_dinv(float* __restrict__ deg) {
    int i = blockIdx.x * blockDim.x + threadIdx.x;
    if (i < NN) {
        float d = deg[i] + 1.f;
        deg[i] = d > 0.f ? rsqrtf(d) : 0.f;
    }
}

// y = dinv^2 * h   (self-loop term; also initializes the accumulator)
__global__ void k_selfloop(const float* __restrict__ h, const float* __restrict__ dinv,
                           float* __restrict__ y, int total) {
    int i = blockIdx.x * blockDim.x + threadIdx.x;
    if (i >= total) return;
    int n = i / DD;
    float di = dinv[n];
    y[i] = di * di * h[i];
}

// one wave per edge: y[col] += (dinv[row]*w*dinv[col]) * h[row]
__global__ void __launch_bounds__(256)
k_scatter(const int* __restrict__ row, const int* __restrict__ col,
          const float* __restrict__ w, const float* __restrict__ dinv,
          const float* __restrict__ h, float* __restrict__ y) {
    int e    = (blockIdx.x * blockDim.x + threadIdx.x) >> 5;
    int lane = threadIdx.x & 31;
    if (e >= EE) return;
    int r  = row[e];
    int cl = col[e];
    float nm = dinv[r] * w[e] * dinv[cl];
    const float* hr = h + (size_t)r * DD;
    float*       yc = y + (size_t)cl * DD;
#pragma unroll
    for (int d = lane; d < DD; d += 32) atomicAdd(&yc[d], nm * hr[d]);
}

// x_next = relu(y + b); bf16 for the next GEMM, optionally f32 branch output
__global__ void k_bias_relu(const float* __restrict__ y, const float* __restrict__ b,
                            __bf16* __restrict__ xbf, float* __restrict__ outf,
                            int write_f32, int total) {
    int i = blockIdx.x * blockDim.x + threadIdx.x;
    if (i >= total) return;
    int d = i % DD;
    float v = y[i] + b[d];
    v = v > 0.f ? v : 0.f;
    xbf[i] = (__bf16)v;
    if (write_f32) outf[i] = v;
}

// softmax attention over the three meta-path embeddings; one wave per node
__global__ void __launch_bounds__(256)
k_attn(const float* __restrict__ o0, const float* __restrict__ o1,
       const float* __restrict__ o2, const float* __restrict__ w_att,
       const float* __restrict__ b_att, float* __restrict__ out) {
    int n    = (blockIdx.x * blockDim.x + threadIdx.x) >> 5;
    int lane = threadIdx.x & 31;
    if (n >= NN) return;
    const float* p0 = o0 + (size_t)n * DD;
    const float* p1 = o1 + (size_t)n * DD;
    const float* p2 = o2 + (size_t)n * DD;
    float s0 = 0.f, s1 = 0.f, s2 = 0.f;
    for (int d = lane; d < DD; d += 32) {
        float wa = w_att[d];
        s0 += p0[d] * wa; s1 += p1[d] * wa; s2 += p2[d] * wa;
    }
#pragma unroll
    for (int off = 16; off; off >>= 1) {
        s0 += __shfl_xor(s0, off, 32);
        s1 += __shfl_xor(s1, off, 32);
        s2 += __shfl_xor(s2, off, 32);
    }
    float ba = b_att[0];
    s0 += ba; s1 += ba; s2 += ba;
    float mx = fmaxf(s0, fmaxf(s1, s2));
    float e0 = __expf(s0 - mx), e1 = __expf(s1 - mx), e2 = __expf(s2 - mx);
    float inv = 1.f / (e0 + e1 + e2);
    e0 *= inv; e1 *= inv; e2 *= inv;
    float* po = out + (size_t)n * DD;
    for (int d = lane; d < DD; d += 32)
        po[d] = e0 * p0[d] + e1 * p1[d] + e2 * p2[d];
}

// ---------------------------------------------------------------------------
static inline int cdiv(long long a, long long b) { return (int)((a + b - 1) / b); }

extern "C" void kernel_launch(void* const* d_in, const int* in_sizes, int n_in,
                              void* d_out, int out_size, void* d_ws, size_t ws_size,
                              hipStream_t stream) {
    (void)in_sizes; (void)n_in; (void)out_size; (void)ws_size;

    const float* init = (const float*)d_in[0];
    const int*   ei[3] = { (const int*)d_in[1], (const int*)d_in[2], (const int*)d_in[3] };
    const float* ea[3] = { (const float*)d_in[4], (const float*)d_in[5], (const float*)d_in[6] };
    const float* pW1 = (const float*)d_in[7];
    const float* pb1 = (const float*)d_in[8];
    const float* pW2 = (const float*)d_in[9];
    const float* pb2 = (const float*)d_in[10];
    const float* Wb[3] = { (const float*)d_in[11], (const float*)d_in[13], (const float*)d_in[15] };
    const float* bb[3] = { (const float*)d_in[12], (const float*)d_in[14], (const float*)d_in[16] };
    const float* w_att = (const float*)d_in[17];
    const float* b_att = (const float*)d_in[18];
    float* out = (float*)d_out;

    // ---- workspace carve-up (256B aligned) ----
    char* ws = (char*)d_ws;
    auto alloc = [&](size_t bytes) -> void* {
        void* p = (void*)ws;
        ws += (bytes + 255) & ~(size_t)255;
        return p;
    };
    __bf16* colT   = (__bf16*)alloc((size_t)DD * MM * sizeof(__bf16));   // B^T for projection
    __bf16* maskbf = (__bf16*)alloc((size_t)NN * MM * sizeof(__bf16));
    float*  rinv   = (float*) alloc((size_t)NN * sizeof(float));
    float*  gemmC  = (float*) alloc((size_t)NN * DD * sizeof(float));    // GEMM output / h
    float*  yacc   = (float*) alloc((size_t)NN * DD * sizeof(float));    // aggregation acc
    __bf16* x0bf   = (__bf16*)alloc((size_t)NN * DD * sizeof(__bf16));   // projection result
    __bf16* xbf    = (__bf16*)alloc((size_t)NN * DD * sizeof(__bf16));   // per-layer act
    __bf16* Wt     = (__bf16*)alloc((size_t)DD * DD * sizeof(__bf16));   // transposed weight
    float*  dinv   = (float*) alloc((size_t)NN * sizeof(float));
    float*  outp[3];
    for (int p = 0; p < 3; ++p) outp[p] = (float*)alloc((size_t)NN * DD * sizeof(float));

    const int ND = NN * DD;
    const int gemm_tiles = (NN / 16) * (DD / 16);      // 20000 waves
    const int gemm_blocks = cdiv((long long)gemm_tiles * 32, 256);

    // ---- projection: col MLP + mask GEMM + masked-mean scale ----
    k_col_proj<<<cdiv(MM, 64), 64, 0, stream>>>(pW1, pb1, pW2, pb2, colT);
    k_mask<<<cdiv((long long)NN * MM, 256), 256, 0, stream>>>(init, maskbf, NN * MM);
    k_rowcnt<<<cdiv((long long)NN * 32, 256), 256, 0, stream>>>(maskbf, rinv);
    k_wmma_gemm<<<gemm_blocks, 256, 0, stream>>>(maskbf, colT, gemmC, MM, DD, gemm_tiles);
    k_scale_bf<<<cdiv(ND, 256), 256, 0, stream>>>(gemmC, rinv, x0bf, ND);

    // ---- three GCN branches ----
    for (int p = 0; p < 3; ++p) {
        const int* erow = ei[p];
        const int* ecol = ei[p] + EE;
        k_zero<<<cdiv(NN, 256), 256, 0, stream>>>(dinv, NN);
        k_deg<<<cdiv(EE, 256), 256, 0, stream>>>(ecol, ea[p], dinv);
        k_dinv<<<cdiv(NN, 256), 256, 0, stream>>>(dinv);
        for (int l = 0; l < 3; ++l) {
            const __bf16* xin = (l == 0) ? x0bf : xbf;
            k_wt<<<cdiv(DD * DD, 256), 256, 0, stream>>>(Wb[p] + (size_t)l * DD * DD, Wt);
            k_wmma_gemm<<<gemm_blocks, 256, 0, stream>>>(xin, Wt, gemmC, DD, DD, gemm_tiles);
            k_selfloop<<<cdiv(ND, 256), 256, 0, stream>>>(gemmC, dinv, yacc, ND);
            k_scatter<<<cdiv((long long)EE * 32, 256), 256, 0, stream>>>(
                erow, ecol, ea[p], dinv, gemmC, yacc);
            k_bias_relu<<<cdiv(ND, 256), 256, 0, stream>>>(
                yacc, bb[p] + (size_t)l * DD, xbf, outp[p], (l == 2) ? 1 : 0, ND);
        }
    }

    // ---- meta-path softmax attention ----
    k_attn<<<cdiv((long long)NN * 32, 256), 256, 0, stream>>>(
        outp[0], outp[1], outp[2], w_att, b_att, out);
}
